// GumbelCodebook_36644660969898
// MI455X (gfx1250) — matrix operand order repaired and checked by reading, not compiled, not run on hardware
//
#include <hip/hip_runtime.h>
#include <hip/hip_bf16.h>
#include <math.h>

typedef __attribute__((ext_vector_type(16))) _Float16 v16h;
typedef __attribute__((ext_vector_type(8)))  _Float16 v8h;
typedef __attribute__((ext_vector_type(8)))  float    v8f;

#define BT_TOTAL 32768      // B*T
#define DIM      1024       // D
#define ECOLS    640        // G*N
#define NGRP     320        // N per group
#define MTILE    32         // rows per workgroup
#define KB       32         // K-chunk (one f16 WMMA K)
#define APAD     40         // halves per A-LDS row (32 data + 8 pad; 80B rows keep b128 align, spread banks)
#define LPAD     641        // floats per logits LDS row (odd stride kills 64-bank conflicts on column walks)

// ---- LDS layout ----
// phase 1: A staging only
#define OFF_AHI   0
#define OFF_ALO   2560                      // 32*APAD*2
// phase 2 union (after K loop): logits + idx/p arrays
#define OFF_IDX   (MTILE*LPAD*4)            // 82048
#define OFF_P     (MTILE*LPAD*4 + 64*4)
#define LDS_BYTES (MTILE*LPAD*4 + 64*4 + 64*4)   // 82560

// ---- workspace layout (d_ws) ----
#define WS_COUNTS 0                         // 640 floats
#define WS_WTHI   4096                      // [640][1024] f16
#define WS_WTLO   (4096 + ECOLS*DIM*2)      // [640][1024] f16
// total ~2.63 MB

__global__ void __launch_bounds__(256) vq_zero_counts(float* counts) {
    int t = blockIdx.x * blockDim.x + threadIdx.x;
    if (t < ECOLS) counts[t] = 0.0f;
}

// Pre-split proj_w into transposed hi/lo f16: WT[e][k].  Reads coalesced over e.
__global__ void __launch_bounds__(128) vq_split_w(const float* __restrict__ w,
                                                  _Float16* __restrict__ wthi,
                                                  _Float16* __restrict__ wtlo) {
    int e = blockIdx.x * 128 + threadIdx.x;      // 0..639
    if (e >= ECOLS) return;
    for (int k = 0; k < DIM; ++k) {
        float v = w[(size_t)k * ECOLS + e];
        _Float16 h = (_Float16)v;
        wthi[(size_t)e * DIM + k] = h;
        wtlo[(size_t)e * DIM + k] = (_Float16)(v - (float)h);
    }
}

__global__ void __launch_bounds__(256, 2) vq_main(
    const float* __restrict__ x,        // [BT,1024]
    const float* __restrict__ gumbel,   // [BT,640]  (btgn flat == row*640 + g*320 + n)
    const float* __restrict__ codebook, // [2,320,512]
    const _Float16* __restrict__ wthi,  // [640][1024] transposed split-f16 W (L2-resident)
    const _Float16* __restrict__ wtlo,
    const float* __restrict__ proj_b,   // [640]
    float* __restrict__ out,            // [BT,1024]
    float* __restrict__ counts)         // [640] histogram of p by (g,idx)
{
    extern __shared__ char smem[];
    _Float16* Ahi = (_Float16*)(smem + OFF_AHI);
    _Float16* Alo = (_Float16*)(smem + OFF_ALO);
    float*    logits = (float*)smem;            // reused after the K loop
    int*      idxArr = (int*)(smem + OFF_IDX);
    float*    pArr   = (float*)(smem + OFF_P);

    const int tid    = threadIdx.x;
    const int wave   = tid >> 5;
    const int lane   = tid & 31;
    const int laneLo = lane & 15;
    const int laneHi = lane >> 4;
    const int rowBase = blockIdx.x * MTILE;

    v8f acc[2][5];
    #pragma unroll
    for (int mi = 0; mi < 2; ++mi)
        #pragma unroll
        for (int ni = 0; ni < 5; ++ni) acc[mi][ni] = {};

    // 16-bit A 16x32 layout: lanes0-15 K 0..15; lanes16-31 K 16..31 (contiguous, matches B)
    const int koffA = laneHi ? 16 : 0;
    // 16-bit B 32x16 layout: lanes0-15 K 0..15; lanes16-31 K 16..31 (contiguous per lane)
    const int koffB = laneHi ? 16 : 0;

    for (int k0 = 0; k0 < DIM; k0 += KB) {
        // ---- cooperative stage: x chunk [32 rows x 32 k] -> split f16 in LDS ----
        #pragma unroll
        for (int i = 0; i < 4; ++i) {
            int flat = i * 256 + tid;             // 1024 elements
            int r  = flat >> 5;
            int kk = flat & 31;
            float v = x[(size_t)(rowBase + r) * DIM + k0 + kk];
            _Float16 h = (_Float16)v;
            Ahi[r * APAD + kk] = h;
            Alo[r * APAD + kk] = (_Float16)(v - (float)h);
        }
        __syncthreads();

        v16h ahi[2], alo[2];
        #pragma unroll
        for (int mi = 0; mi < 2; ++mi) {
            int r = mi * 16 + laneLo;
            ((v8h*)&ahi[mi])[0] = *(const v8h*)&Ahi[r * APAD + koffA];
            ((v8h*)&ahi[mi])[1] = *(const v8h*)&Ahi[r * APAD + 8 + koffA];
            ((v8h*)&alo[mi])[0] = *(const v8h*)&Alo[r * APAD + koffA];
            ((v8h*)&alo[mi])[1] = *(const v8h*)&Alo[r * APAD + 8 + koffA];
        }
        #pragma unroll
        for (int ni = 0; ni < 5; ++ni) {
            int n = (wave * 5 + ni) * 16 + laneLo;
            size_t boff = (size_t)n * DIM + k0 + koffB;   // 32B-aligned, 32 contiguous halves
            v16h bhi = *(const v16h*)(wthi + boff);
            v16h blo = *(const v16h*)(wtlo + boff);
            #pragma unroll
            for (int mi = 0; mi < 2; ++mi) {
                acc[mi][ni] = __builtin_amdgcn_wmma_f32_16x16x32_f16(
                    false, ahi[mi], false, bhi, (short)0, acc[mi][ni], false, false);
                acc[mi][ni] = __builtin_amdgcn_wmma_f32_16x16x32_f16(
                    false, ahi[mi], false, blo, (short)0, acc[mi][ni], false, false);
                acc[mi][ni] = __builtin_amdgcn_wmma_f32_16x16x32_f16(
                    false, alo[mi], false, bhi, (short)0, acc[mi][ni], false, false);
            }
        }
        __syncthreads();
    }

    // ---- spill logits (+bias+gumbel) into LDS (C layout: VGPR v -> M = v + 8*laneHi, N = laneLo) ----
    #pragma unroll
    for (int ni = 0; ni < 5; ++ni) {
        int col = (wave * 5 + ni) * 16 + laneLo;
        float bias = proj_b[col];
        #pragma unroll
        for (int mi = 0; mi < 2; ++mi) {
            #pragma unroll
            for (int v = 0; v < 8; ++v) {
                int rl = mi * 16 + v + laneHi * 8;
                float gn = gumbel[(size_t)(rowBase + rl) * ECOLS + col];
                logits[rl * LPAD + col] = acc[mi][ni][v] + bias + gn;
            }
        }
    }
    __syncthreads();

    // ---- per-(row,group) online softmax: argmax + sum_exp -> p ----
    if (tid < 64) {
        int row = tid >> 1;
        int g   = tid & 1;
        const float* lp = &logits[row * LPAD + g * NGRP];
        float m = lp[0];
        float s = 1.0f;
        int best = 0;
        for (int n = 1; n < NGRP; ++n) {
            float v = lp[n];
            if (v > m) { s = s * __expf(m - v) + 1.0f; m = v; best = n; }
            else       { s += __expf(v - m); }
        }
        float ys = 1.0f / s;                   // y_soft at the argmax
        float p  = (1.0f - ys) + ys;           // matches (y_hard - sg(y_soft) + y_soft)
        idxArr[tid] = best;
        pArr[tid]   = p;
        atomicAdd(&counts[g * NGRP + best], p);
    }
    __syncthreads();

    // ---- gather codebook rows, scale by p, write output (float4 coalesced) ----
    #pragma unroll 4
    for (int i = 0; i < 32; ++i) {
        int flat4 = i * 256 + tid;           // 8192 float4 = 32 rows * 1024 floats
        int row = flat4 >> 8;
        int c4  = flat4 & 255;
        int col = c4 << 2;
        int g   = col >> 9;
        int d   = col & 511;
        int idx   = idxArr[row * 2 + g];
        float p   = pArr[row * 2 + g];
        const float4 cb = *(const float4*)&codebook[((size_t)g * NGRP + idx) * 512 + d];
        float4 o; o.x = p * cb.x; o.y = p * cb.y; o.z = p * cb.z; o.w = p * cb.w;
        *(float4*)&out[(size_t)(rowBase + row) * DIM + col] = o;
    }
}

__global__ void __launch_bounds__(256) vq_diversity(const float* __restrict__ counts,
                                                    float* __restrict__ out_loss) {
    __shared__ float red[256];
    float a = 0.0f;
    for (int i = threadIdx.x; i < ECOLS; i += 256) {
        float avg = counts[i] * (1.0f / 32768.0f);
        a += -avg * logf(avg + 1e-7f);
    }
    red[threadIdx.x] = a;
    __syncthreads();
    for (int s = 128; s > 0; s >>= 1) {
        if (threadIdx.x < s) red[threadIdx.x] += red[threadIdx.x + s];
        __syncthreads();
    }
    if (threadIdx.x == 0) out_loss[0] = red[0] * 0.5f;   // divide by G=2
}

extern "C" void kernel_launch(void* const* d_in, const int* in_sizes, int n_in,
                              void* d_out, int out_size, void* d_ws, size_t ws_size,
                              hipStream_t stream) {
    const float* x        = (const float*)d_in[0];
    const float* gumbel   = (const float*)d_in[1];
    const float* codebook = (const float*)d_in[2];
    const float* proj_w   = (const float*)d_in[3];
    const float* proj_b   = (const float*)d_in[4];
    float* out    = (float*)d_out;

    char* ws = (char*)d_ws;
    float*    counts = (float*)(ws + WS_COUNTS);
    _Float16* wthi   = (_Float16*)(ws + WS_WTHI);
    _Float16* wtlo   = (_Float16*)(ws + WS_WTLO);

    (void)hipFuncSetAttribute((const void*)vq_main,
                              hipFuncAttributeMaxDynamicSharedMemorySize, LDS_BYTES);

    vq_zero_counts<<<3, 256, 0, stream>>>(counts);
    vq_split_w<<<5, 128, 0, stream>>>(proj_w, wthi, wtlo);
    vq_main<<<BT_TOTAL / MTILE, 256, LDS_BYTES, stream>>>(
        x, gumbel, codebook, wthi, wtlo, proj_b, out, counts);
    vq_diversity<<<1, 256, 0, stream>>>(counts, out + (size_t)BT_TOTAL * DIM);
}
